// MultiHeadedAttention_15410342658486
// MI455X (gfx1250) — compile-verified, matrix operand order checked by reading, not run on hardware
//
#include <hip/hip_runtime.h>
#include <hip/hip_bf16.h>

typedef __bf16 bf16;
typedef __attribute__((ext_vector_type(16))) __bf16 v16bf;
typedef __attribute__((ext_vector_type(8)))  __bf16 v8bf;
typedef __attribute__((ext_vector_type(8)))  float  v8f;

#define D_EMBED 1024
#define SEQ     2048
#define BATCH   4
#define NHEAD   16
#define HD      64
#define NROWS   (BATCH * SEQ)   // 8192

// ---------------------------------------------------------------------------
// gfx1250 async global->LDS copy (GLOBAL_LOAD_ASYNC_TO_LDS_B128, ASYNCcnt),
// via inline asm (ISA 15.18.3 opcode 98, GV addressing: vdst = LDS byte
// address VGPR, vaddr = 64-bit global address VGPR pair).
// ---------------------------------------------------------------------------
#if defined(__gfx1250__)
#define HAVE_ASYNC_LDS 1
#endif

__device__ __forceinline__ void async_cp16(const void* g, void* l) {
#ifdef HAVE_ASYNC_LDS
  // Flat->LDS aperture rule: LDS address = low 32 bits of the flat address (ISA 10.2).
  unsigned lds_off = (unsigned)(unsigned long long)l;
  unsigned long long gaddr = (unsigned long long)g;
  asm volatile("global_load_async_to_lds_b128 %0, %1, off"
               :: "v"(lds_off), "v"(gaddr)
               : "memory");
#else
  *(uint4*)l = *(const uint4*)g;
#endif
}

__device__ __forceinline__ void async_wait0() {
#ifdef HAVE_ASYNC_LDS
#if defined(__has_builtin)
#if __has_builtin(__builtin_amdgcn_s_wait_asynccnt)
#define USED_WAIT_BUILTIN 1
  __builtin_amdgcn_s_wait_asynccnt(0);
#endif
#endif
#ifndef USED_WAIT_BUILTIN
  asm volatile("s_wait_asynccnt 0x0" ::: "memory");
#endif
#endif
}

// ---------------------------------------------------------------------------
// WMMA helpers (CDNA5 wave32, v_wmma_f32_16x16x32_bf16)
// ---------------------------------------------------------------------------
__device__ __forceinline__ v8f wmma_bf16(v16bf a, v16bf b, v8f c) {
  // (neg_a, A, neg_b, B, c_mod, C, reuse_a, reuse_b)
  return __builtin_amdgcn_wmma_f32_16x16x32_bf16(false, a, false, b,
                                                 (short)0, c, false, false);
}

// Load a 16x32 bf16 A- or B-fragment from a row-major, K-contiguous buffer.
// ISA 7.12.2 layout: lane&15 = row (M for A / N for B); lanes>=16 hold the
// K+8 group; element e -> K = kb + e (e<8) or kb + 16 + (e-8).
__device__ __forceinline__ v16bf load_frag(const bf16* p) {
  v8bf lo = *(const v8bf*)(p);        // K = kb + 0..7
  v8bf hi = *(const v8bf*)(p + 16);   // K = kb + 16..23
  v16bf f;
#pragma unroll
  for (int i = 0; i < 8; ++i) { f[i] = lo[i]; f[i + 8] = hi[i]; }
  return f;
}

__device__ __forceinline__ v16bf load_frag_rm(const bf16* base, int row0,
                                              int col0, int ld) {
  const int lane = threadIdx.x & 31;
  return load_frag(base + (size_t)(row0 + (lane & 15)) * ld + col0 +
                   ((lane >> 4) << 3));
}

// ---------------------------------------------------------------------------
// Kernel 1: Y = X * W^T (fp32 in, bf16 out).
// Block: 256 thr = 8 waves; block tile 128(M) x 128(N); wave tile 32x64;
// K-step 32 -> 8 WMMAs per wave per K-step against 6 fragment loads.
// ---------------------------------------------------------------------------
#define T1_M 128
#define T1_N 128
#define T1_K 32
#define LDA1 (T1_K + 8)   // 40 bf16 = 80 B rows, 16B aligned

template <int TRANSPOSE_OUT>
__global__ __launch_bounds__(256)
void qkv_proj_kernel(const float* __restrict__ X, const float* __restrict__ W,
                     bf16* __restrict__ Out, float scale) {
  __shared__ bf16 As[T1_M][LDA1];
  __shared__ bf16 Bs[T1_N][LDA1];

  const int tid = threadIdx.x;
  const int m0 = blockIdx.x * T1_M;
  const int n0 = blockIdx.y * T1_N;
  const int wave = tid >> 5;
  const int wm = (wave & 3) * 32;   // 4 waves along M
  const int wn = (wave >> 2) * 64;  // 2 waves along N

  v8f acc[2][4] = {};

  const int srow = tid >> 1, skseg = (tid & 1) * 16;  // 2 thr/row, 16 f32 each
  const float* gA = X + (size_t)(m0 + srow) * D_EMBED + skseg;
  const float* gB = W + (size_t)(n0 + srow) * D_EMBED + skseg;

  for (int kt = 0; kt < D_EMBED; kt += T1_K) {
    if (kt + T1_K < D_EMBED) {  // prefetch next K tile (global_prefetch_b8)
      __builtin_prefetch(gA + kt + T1_K, 0, 0);
      __builtin_prefetch(gB + kt + T1_K, 0, 0);
    }
    {  // stage A: 128x32 fp32 -> bf16 LDS
      const float4* gp = (const float4*)(gA + kt);
      float4 f0 = gp[0], f1 = gp[1], f2 = gp[2], f3 = gp[3];
      v8bf t0, t1;
      t0[0] = (bf16)f0.x; t0[1] = (bf16)f0.y; t0[2] = (bf16)f0.z; t0[3] = (bf16)f0.w;
      t0[4] = (bf16)f1.x; t0[5] = (bf16)f1.y; t0[6] = (bf16)f1.z; t0[7] = (bf16)f1.w;
      t1[0] = (bf16)f2.x; t1[1] = (bf16)f2.y; t1[2] = (bf16)f2.z; t1[3] = (bf16)f2.w;
      t1[4] = (bf16)f3.x; t1[5] = (bf16)f3.y; t1[6] = (bf16)f3.z; t1[7] = (bf16)f3.w;
      *(v8bf*)&As[srow][skseg] = t0;
      *(v8bf*)&As[srow][skseg + 8] = t1;
    }
    {  // stage B: 128x32 fp32 -> bf16 LDS (W row n = output column n)
      const float4* gp = (const float4*)(gB + kt);
      float4 f0 = gp[0], f1 = gp[1], f2 = gp[2], f3 = gp[3];
      v8bf t0, t1;
      t0[0] = (bf16)f0.x; t0[1] = (bf16)f0.y; t0[2] = (bf16)f0.z; t0[3] = (bf16)f0.w;
      t0[4] = (bf16)f1.x; t0[5] = (bf16)f1.y; t0[6] = (bf16)f1.z; t0[7] = (bf16)f1.w;
      t1[0] = (bf16)f2.x; t1[1] = (bf16)f2.y; t1[2] = (bf16)f2.z; t1[3] = (bf16)f2.w;
      t1[4] = (bf16)f3.x; t1[5] = (bf16)f3.y; t1[6] = (bf16)f3.z; t1[7] = (bf16)f3.w;
      *(v8bf*)&Bs[srow][skseg] = t0;
      *(v8bf*)&Bs[srow][skseg + 8] = t1;
    }
    __syncthreads();

    v16bf a0 = load_frag_rm(&As[0][0], wm, 0, LDA1);
    v16bf a1 = load_frag_rm(&As[0][0], wm + 16, 0, LDA1);
#pragma unroll
    for (int ni = 0; ni < 4; ++ni) {
      v16bf bn = load_frag_rm(&Bs[0][0], wn + ni * 16, 0, LDA1);
      acc[0][ni] = wmma_bf16(a0, bn, acc[0][ni]);
      acc[1][ni] = wmma_bf16(a1, bn, acc[1][ni]);
    }
    __syncthreads();
  }

  // Epilogue: C layout = VGPR i: lanes 0-15 row M=i, lanes 16-31 row M=i+8.
  const int lane = tid & 31;
  const int chi = lane >> 4, nlo = lane & 15;
#pragma unroll
  for (int mi = 0; mi < 2; ++mi)
#pragma unroll
    for (int ni = 0; ni < 4; ++ni)
#pragma unroll
      for (int i = 0; i < 8; ++i) {
        int m = m0 + wm + mi * 16 + i + chi * 8;
        int n = n0 + wn + ni * 16 + nlo;
        int b = m >> 11, s = m & (SEQ - 1);
        int h = n >> 6, hd = n & (HD - 1);
        size_t off;
        if (TRANSPOSE_OUT)
          off = (((size_t)b * NHEAD + h) * HD + hd) * SEQ + s;   // [B,H,Hd,S]
        else
          off = (((size_t)b * NHEAD + h) * SEQ + s) * HD + hd;   // [B,H,S,Hd]
        Out[off] = (bf16)(acc[mi][ni][i] * scale);
      }
}

// ---------------------------------------------------------------------------
// Kernel 2: causal flash attention. One block = (b,h) x 64-query tile.
// 128 thr = 4 waves; wave handles 16 query rows; keys streamed in 64-tiles
// staged into LDS with GLOBAL_LOAD_ASYNC_TO_LDS_B128 (ASYNCcnt).
// ---------------------------------------------------------------------------
#define Q_TILE 64
#define K_TILE 64
#define LDK (HD + 8)       // 72 bf16 = 144 B rows, 16B aligned
#define LDV (K_TILE + 8)   // 72

__global__ __launch_bounds__(128)
void attn_kernel(const bf16* __restrict__ Q, const bf16* __restrict__ Kmat,
                 const bf16* __restrict__ Vt, bf16* __restrict__ Aout) {
  __shared__ bf16 Ks[K_TILE][LDK];     // keys:   [j][hd]
  __shared__ bf16 Vs[HD][LDV];         // V^T:    [hd][j]
  __shared__ bf16 Ps[4][16][LDV];      // per-wave P (C->A relayout)

  const int tid = threadIdx.x;
  const int lane = tid & 31;
  const int w = tid >> 5;
  const int bh = blockIdx.x;
  const int b = bh >> 4, h = bh & 15;
  const int q0 = blockIdx.y * Q_TILE;
  const int qw = q0 + w * 16;
  const int chi = lane >> 4, nlo = lane & 15;

  const bf16* Qbh = Q + (size_t)bh * SEQ * HD;
  const bf16* Kbh = Kmat + (size_t)bh * SEQ * HD;
  const bf16* Vbh = Vt + (size_t)bh * HD * SEQ;

  // Q fragments (pre-scaled by 1/sqrt(Hd)): reused for every key tile.
  v16bf aq0 = load_frag_rm(Qbh, qw, 0, HD);
  v16bf aq1 = load_frag_rm(Qbh, qw, 32, HD);

  v8f o[4] = {};
  float mr[8], lr[8];
#pragma unroll
  for (int i = 0; i < 8; ++i) { mr[i] = -1e30f; lr[i] = 0.f; }

  const int srow = tid >> 1, sseg = (tid & 1) * 32;  // staging: 2 thr/row

  for (int j0 = 0; j0 < q0 + Q_TILE; j0 += K_TILE) {
    {  // stage K tile (64x64 bf16): async copy, 4x16B per thread
      const bf16* gp = Kbh + (size_t)(j0 + srow) * HD + sseg;
      bf16* dp = &Ks[srow][sseg];
#pragma unroll
      for (int c = 0; c < 4; ++c) async_cp16(gp + 8 * c, dp + 8 * c);
    }
    {  // stage V^T tile (64x64 bf16)
      const bf16* gp = Vbh + (size_t)srow * SEQ + j0 + sseg;
      bf16* dp = &Vs[srow][sseg];
#pragma unroll
      for (int c = 0; c < 4; ++c) async_cp16(gp + 8 * c, dp + 8 * c);
    }
    async_wait0();     // s_wait_asynccnt 0: our transfers landed in LDS
    __syncthreads();   // everyone else's too

    // S = Q K^T : 16 x 64, 8 WMMAs
    v8f s[4] = {};
#pragma unroll
    for (int n = 0; n < 4; ++n) {
      v16bf bk0 = load_frag_rm(&Ks[0][0], n * 16, 0, LDK);
      v16bf bk1 = load_frag_rm(&Ks[0][0], n * 16, 32, LDK);
      s[n] = wmma_bf16(aq0, bk0, s[n]);
      s[n] = wmma_bf16(aq1, bk1, s[n]);
    }

    if (j0 + K_TILE - 1 > qw) {  // causal mask on (partially) diagonal tiles
#pragma unroll
      for (int n = 0; n < 4; ++n)
#pragma unroll
        for (int i = 0; i < 8; ++i) {
          int col = j0 + n * 16 + nlo;
          int qr = qw + i + chi * 8;
          if (col > qr) s[n][i] = -1e30f;
        }
    }

    // Online softmax: row stats replicated across each 16-lane half.
    float mt[8];
#pragma unroll
    for (int i = 0; i < 8; ++i)
      mt[i] = fmaxf(fmaxf(s[0][i], s[1][i]), fmaxf(s[2][i], s[3][i]));
#pragma unroll
    for (int xm = 1; xm <= 8; xm <<= 1)
#pragma unroll
      for (int i = 0; i < 8; ++i)
        mt[i] = fmaxf(mt[i], __shfl_xor(mt[i], xm, 32));

    float sc[8], rs[8];
#pragma unroll
    for (int i = 0; i < 8; ++i) {
      float mn = fmaxf(mr[i], mt[i]);
      sc[i] = __expf(mr[i] - mn);
      mr[i] = mn;
      rs[i] = 0.f;
    }
#pragma unroll
    for (int n = 0; n < 4; ++n)
#pragma unroll
      for (int i = 0; i < 8; ++i) {
        float p = __expf(s[n][i] - mr[i]);
        s[n][i] = p;
        rs[i] += p;
      }
#pragma unroll
    for (int xm = 1; xm <= 8; xm <<= 1)
#pragma unroll
      for (int i = 0; i < 8; ++i) rs[i] += __shfl_xor(rs[i], xm, 32);
#pragma unroll
    for (int i = 0; i < 8; ++i) lr[i] = lr[i] * sc[i] + rs[i];
#pragma unroll
    for (int n = 0; n < 4; ++n)
#pragma unroll
      for (int i = 0; i < 8; ++i) o[n][i] *= sc[i];

    // P: C-layout -> row-major LDS (wave-private) -> A-fragments
#pragma unroll
    for (int n = 0; n < 4; ++n)
#pragma unroll
      for (int i = 0; i < 8; ++i)
        Ps[w][i + chi * 8][n * 16 + nlo] = (bf16)s[n][i];

    v16bf ap0 = load_frag_rm(&Ps[w][0][0], 0, 0, LDV);
    v16bf ap1 = load_frag_rm(&Ps[w][0][0], 0, 32, LDV);
#pragma unroll
    for (int d = 0; d < 4; ++d) {  // O += P V : 8 WMMAs
      v16bf bv0 = load_frag_rm(&Vs[0][0], d * 16, 0, LDV);
      v16bf bv1 = load_frag_rm(&Vs[0][0], d * 16, 32, LDV);
      o[d] = wmma_bf16(ap0, bv0, o[d]);
      o[d] = wmma_bf16(ap1, bv1, o[d]);
    }
    __syncthreads();
  }

  float inv[8];
#pragma unroll
  for (int i = 0; i < 8; ++i) inv[i] = 1.0f / lr[i];

  // Write attention output in [B, S, H*Hd] so out-proj reads K-contiguous.
  bf16* Ob = Aout + ((size_t)b * SEQ) * D_EMBED + (size_t)h * HD;
#pragma unroll
  for (int d = 0; d < 4; ++d)
#pragma unroll
    for (int i = 0; i < 8; ++i) {
      int qr = qw + i + chi * 8;
      Ob[(size_t)qr * D_EMBED + d * 16 + nlo] = (bf16)(o[d][i] * inv[i]);
    }
}

// ---------------------------------------------------------------------------
// Kernel 3: Out = A * Wo^T (bf16 A from workspace, fp32 output).
// Same 128x128 block / 32x64 wave tiling as kernel 1.
// ---------------------------------------------------------------------------
__global__ __launch_bounds__(256)
void out_proj_kernel(const bf16* __restrict__ A, const float* __restrict__ W,
                     float* __restrict__ Out) {
  __shared__ bf16 As[T1_M][LDA1];
  __shared__ bf16 Bs[T1_N][LDA1];

  const int tid = threadIdx.x;
  const int m0 = blockIdx.x * T1_M;
  const int n0 = blockIdx.y * T1_N;
  const int wave = tid >> 5;
  const int wm = (wave & 3) * 32;
  const int wn = (wave >> 2) * 64;

  v8f acc[2][4] = {};

  const int srow = tid >> 1, skseg = (tid & 1) * 16;
  const bf16*  gA = A + (size_t)(m0 + srow) * D_EMBED + skseg;
  const float* gB = W + (size_t)(n0 + srow) * D_EMBED + skseg;

  for (int kt = 0; kt < D_EMBED; kt += T1_K) {
    if (kt + T1_K < D_EMBED) {
      __builtin_prefetch(gA + kt + T1_K, 0, 0);
      __builtin_prefetch(gB + kt + T1_K, 0, 0);
    }
    {  // stage A: already bf16, straight 2x16B copy
      const uint4* gp = (const uint4*)(gA + kt);
      uint4 u0 = gp[0], u1 = gp[1];
      *(uint4*)&As[srow][skseg] = u0;
      *(uint4*)&As[srow][skseg + 8] = u1;
    }
    {  // stage B: fp32 W -> bf16
      const float4* gp = (const float4*)(gB + kt);
      float4 f0 = gp[0], f1 = gp[1], f2 = gp[2], f3 = gp[3];
      v8bf t0, t1;
      t0[0] = (bf16)f0.x; t0[1] = (bf16)f0.y; t0[2] = (bf16)f0.z; t0[3] = (bf16)f0.w;
      t0[4] = (bf16)f1.x; t0[5] = (bf16)f1.y; t0[6] = (bf16)f1.z; t0[7] = (bf16)f1.w;
      t1[0] = (bf16)f2.x; t1[1] = (bf16)f2.y; t1[2] = (bf16)f2.z; t1[3] = (bf16)f2.w;
      t1[4] = (bf16)f3.x; t1[5] = (bf16)f3.y; t1[6] = (bf16)f3.z; t1[7] = (bf16)f3.w;
      *(v8bf*)&Bs[srow][skseg] = t0;
      *(v8bf*)&Bs[srow][skseg + 8] = t1;
    }
    __syncthreads();

    v16bf a0 = load_frag_rm(&As[0][0], wm, 0, LDA1);
    v16bf a1 = load_frag_rm(&As[0][0], wm + 16, 0, LDA1);
#pragma unroll
    for (int ni = 0; ni < 4; ++ni) {
      v16bf bn = load_frag_rm(&Bs[0][0], wn + ni * 16, 0, LDA1);
      acc[0][ni] = wmma_bf16(a0, bn, acc[0][ni]);
      acc[1][ni] = wmma_bf16(a1, bn, acc[1][ni]);
    }
    __syncthreads();
  }

  const int lane = tid & 31;
  const int chi = lane >> 4, nlo = lane & 15;
#pragma unroll
  for (int mi = 0; mi < 2; ++mi)
#pragma unroll
    for (int ni = 0; ni < 4; ++ni)
#pragma unroll
      for (int i = 0; i < 8; ++i) {
        int m = m0 + wm + mi * 16 + i + chi * 8;
        int n = n0 + wn + ni * 16 + nlo;
        Out[(size_t)m * D_EMBED + n] = acc[mi][ni][i];
      }
}

// ---------------------------------------------------------------------------
extern "C" void kernel_launch(void* const* d_in, const int* in_sizes, int n_in,
                              void* d_out, int out_size, void* d_ws,
                              size_t ws_size, hipStream_t stream) {
  const float* x  = (const float*)d_in[0];
  const float* Wq = (const float*)d_in[1];
  const float* Wk = (const float*)d_in[2];
  const float* Wv = (const float*)d_in[3];
  const float* Wo = (const float*)d_in[4];
  float* out = (float*)d_out;

  const size_t mat = (size_t)NROWS * D_EMBED;  // 8,388,608 elems (16 MB bf16)
  bf16* Qb = (bf16*)d_ws;                      // [B,H,S,Hd], pre-scaled
  bf16* Kb = Qb + mat;                         // [B,H,S,Hd]
  bf16* Vt = Kb + mat;                         // [B,H,Hd,S]
  bf16* At = Vt + mat;                         // [B,S,D]

  dim3 gProj(NROWS / T1_M, D_EMBED / T1_N);    // 64 x 8
  qkv_proj_kernel<0><<<gProj, 256, 0, stream>>>(x, Wq, Qb, 0.125f);
  qkv_proj_kernel<0><<<gProj, 256, 0, stream>>>(x, Wk, Kb, 1.0f);
  qkv_proj_kernel<1><<<gProj, 256, 0, stream>>>(x, Wv, Vt, 1.0f);

  dim3 gAttn(BATCH * NHEAD, SEQ / Q_TILE);     // 64 x 32
  attn_kernel<<<gAttn, 128, 0, stream>>>(Qb, Kb, Vt, At);

  out_proj_kernel<<<gProj, 256, 0, stream>>>(At, Wo, out);
}
